// MBSeederSLFCAMS_32736240730421
// MI455X (gfx1250) — matrix operand order repaired and checked by reading, not compiled, not run on hardware
//
#include <hip/hip_runtime.h>
#include <hip/hip_bf16.h>

// ---------------------------------------------------------------------------
// MBSeeder pipeline for MI455X (gfx1250, wave32).
//
// Memory-bound problem (64MB in / 64MB out, fits in 192MB L2).  The 11x11
// binary erosion is computed as a separable box-sum via V_WMMA_F32_16X16X32_F16
// against constant banded 0/1 matrices (band fragments are shift-invariant and
// built per-lane in registers).  The column pass stages its 64-row rowSum halo
// into LDS with GLOBAL_LOAD_ASYNC_TO_LDS_B128 (ASYNCcnt-tracked async copy).
// All selection thresholds are derived from commutative histogram counts only
// -> fully deterministic across graph replays.
// ---------------------------------------------------------------------------

typedef __attribute__((ext_vector_type(16))) _Float16 v16h;
typedef __attribute__((ext_vector_type(8)))  float    v8f;

union F16x16 { v16h v; _Float16 e[16]; };
union F32x8  { v8f  v; float    e[8];  };

#define IMG_HW 65536
#define IMG_W  256
#define K_FG   100u
#define K_BG   6553u   // int(0.1 * 256 * 256)

// ---- deterministic per-pixel hash "random" scores (16-bit keys) -----------
__device__ __forceinline__ unsigned mix32(unsigned x) {
  x ^= x >> 16; x *= 0x7feb352du;
  x ^= x >> 15; x *= 0x846ca68bu;
  x ^= x >> 16; return x;
}
__device__ __forceinline__ int keyF(int img, int p) {
  return (int)(mix32((((unsigned)img << 16) ^ (unsigned)p) * 0x9E3779B9u + 0x12345678u) >> 16);
}
__device__ __forceinline__ int keyB(int img, int p) {
  return (int)(mix32((((unsigned)img << 16) ^ (unsigned)p) * 0x85EBCA6Bu + 0xBEEF1234u) >> 16);
}
// 16-bit monotone key for cam; high byte == clip(floor(cam*255)) (Otsu bin)
__device__ __forceinline__ int camKey16(float cam) {
  int k = (int)floorf(cam * 65280.0f);   // 255*256
  return k < 0 ? 0 : (k > 65535 ? 65535 : k);
}

// ---------------------------------------------------------------------------
// 1) per-image 256-bin histogram of floor(cam*255) + bad flag (min==max)
// ---------------------------------------------------------------------------
__global__ __launch_bounds__(256) void hist_kernel(const float* __restrict__ x,
                                                   unsigned* __restrict__ histWS,
                                                   int* __restrict__ badWS) {
  __shared__ unsigned h[256];
  __shared__ int mn, mx;
  const int img = blockIdx.x, t = threadIdx.x;
  h[t] = 0u;
  if (t == 0) { mn = 256; mx = -1; }
  __syncthreads();
  const float* cam = x + (size_t)img * IMG_HW;
  int lmn = 256, lmx = -1;
  for (int p = t; p < IMG_HW; p += 256) {
    if (p + 4096 < IMG_HW) __builtin_prefetch(cam + p + 4096, 0, 1);  // global_prefetch_b8
    int ci = camKey16(cam[p]) >> 8;
    atomicAdd(&h[ci], 1u);
    lmn = min(lmn, ci); lmx = max(lmx, ci);
  }
  atomicMin(&mn, lmn); atomicMax(&mx, lmx);
  __syncthreads();
  histWS[img * 256 + t] = h[t];
  if (t == 0) badWS[img] = (mn == mx) ? 1 : 0;
}

// ---------------------------------------------------------------------------
// 2) Otsu threshold per image (one thread per image; 256 bins is tiny)
// ---------------------------------------------------------------------------
__global__ __launch_bounds__(256) void otsu_kernel(const unsigned* __restrict__ histWS,
                                                   float* __restrict__ thWS) {
  const int img = threadIdx.x;          // grid = 1 block of 256 threads
  const unsigned* hu = histWS + img * 256;
  float total = 0.f, sTot = 0.f;
  for (int i = 0; i < 256; ++i) { float hv = (float)hu[i]; total += hv; sTot += hv * (float)i; }
  float w1 = 0.f, s1 = 0.f, best = -1.f; int bi = 0;
  for (int t = 0; t < 255; ++t) {
    float hv = (float)hu[t];
    w1 += hv; s1 += hv * (float)t;
    float w2 = total - w1;              // w2[t+1]
    float s2 = sTot  - s1;              // s2[t+1]
    float m1 = s1 / fmaxf(w1, 1.f);
    float m2 = s2 / fmaxf(w2, 1.f);
    float dm = m1 - m2;
    float v = w1 * w2 * dm * dm;
    if (v > best) { best = v; bi = t; } // first max, like jnp.argmax
  }
  float th = (float)bi;
  if (th <= 0.f) th = 1.f; else if (th >= 255.f) th = 254.f;
  thWS[img] = th;
}

// ---------------------------------------------------------------------------
// 3) BG candidate cut: 16-bit key such that {camKey16 <= cut} ~ 6553 lowest
//    (two-level deterministic refinement: coarse bin from hist + sub-bin)
// ---------------------------------------------------------------------------
__global__ __launch_bounds__(256) void bgcut_kernel(const float* __restrict__ x,
                                                    const unsigned* __restrict__ histWS,
                                                    int* __restrict__ bgCutWS) {
  __shared__ unsigned h2[256];
  __shared__ int sC, sRem;
  const int img = blockIdx.x, t = threadIdx.x;
  if (t == 0) {
    const unsigned* h = histWS + img * 256;
    unsigned run = 0, below = 0; int C = 255;
    for (int b = 0; b < 256; ++b) { run += h[b]; if (run >= K_BG) { C = b; below = run - h[b]; break; } }
    sC = C; sRem = (int)(K_BG - below);
  }
  h2[t] = 0u;
  __syncthreads();
  const float* cam = x + (size_t)img * IMG_HW;
  for (int p = t; p < IMG_HW; p += 256) {
    int k = camKey16(cam[p]);
    if ((k >> 8) == sC) atomicAdd(&h2[k & 255], 1u);
  }
  __syncthreads();
  if (t == 0) {
    unsigned run = 0; int c2 = 255;
    for (int s = 0; s < 256; ++s) { run += h2[s]; if (run >= (unsigned)sRem) { c2 = s; break; } }
    bgCutWS[img] = (sC << 8) | c2;
  }
}

// ---------------------------------------------------------------------------
// 4) Erosion row pass via WMMA: rowSum[r][c] = sum_{|d|<=5} roi[r][c+d]
//    RowSumTile(16x16) = Roi(16x32)xBand0 + Roi(16x32)xBand1
//    Band fragments are shift-invariant -> computed analytically per lane.
//    rowSum is written as f16 (integers <= 11, exact) so the column pass can
//    async-copy it byte-exact into LDS.
// ---------------------------------------------------------------------------
__global__ __launch_bounds__(256) void rowpass_kernel(const float* __restrict__ x,
                                                      const float* __restrict__ thWS,
                                                      _Float16* __restrict__ rowSum) {
  extern __shared__ _Float16 roiLds[];            // [16][304]: 16-col left pad, 32-col right pad
  const int stripe = blockIdx.x, img = blockIdx.y;
  const int r0 = stripe * 16;
  const float th = thWS[img];
  const float* cam = x + (size_t)img * IMG_HW;
  for (int i = threadIdx.x; i < 16 * 304; i += 256) {
    int r = i / 304, c = i % 304;
    _Float16 v = (_Float16)0.0f;
    if (c >= 16 && c < 272) {                     // global col = c-16 in [0,256)
      float cv = cam[(r0 + r) * IMG_W + (c - 16)];
      v = (floorf(cv * 255.0f) > th) ? (_Float16)1.0f : (_Float16)0.0f;
    }
    roiLds[i] = v;
  }
  __syncthreads();

  const int lane = threadIdx.x & 31, wid = threadIdx.x >> 5;
  const int n = lane & 15, half = lane >> 4;      // B-col / D-col = n ; lane half
  // constant banded B fragments (32x16, K = (e<8?e:e+8) + 8*half):
  //   block0 covers global cols j0-16..j0+15 : nonzero iff k in [n+11, n+21]
  //   block1 covers global cols j0+16..j0+47 : nonzero iff k in [n-21, n-11]
  F16x16 b0, b1;
#pragma unroll
  for (int e = 0; e < 16; ++e) {
    int k = ((e < 8) ? e : e + 8) + 8 * half;
    b0.e[e] = ((unsigned)(k - n - 11) <= 10u) ? (_Float16)1.0f : (_Float16)0.0f;
    b1.e[e] = ((unsigned)(k - n + 21) <= 10u) ? (_Float16)1.0f : (_Float16)0.0f;
  }
  const int m = lane & 15;                        // A row
  for (int tile = wid * 2; tile < wid * 2 + 2; ++tile) {  // 8 waves x 2 tiles = 16 col tiles
    const int j0 = tile * 16;
    F16x16 a0, a1;                                // A 16x32: K = (e<8?e:e+8) + 8*half
#pragma unroll
    for (int e = 0; e < 16; ++e) {
      int K = ((e < 8) ? e : e + 8) + 8 * half;
      a0.e[e] = roiLds[m * 304 + j0 + K];         // padded idx of global col j0-16+K
      a1.e[e] = roiLds[m * 304 + j0 + 32 + K];    // padded idx of global col j0+16+K
    }
    v8f c = {};
    c = __builtin_amdgcn_wmma_f32_16x16x32_f16(false, a0.v, false, b0.v, (short)0, c, false, false);
    c = __builtin_amdgcn_wmma_f32_16x16x32_f16(false, a1.v, false, b1.v, (short)0, c, false, false);
    F32x8 d; d.v = c;
#pragma unroll
    for (int i = 0; i < 8; ++i) {                 // D: row = i + 8*half, col = j0 + n
      int row = r0 + 8 * half + i;
      int col = j0 + n;
      rowSum[(size_t)img * IMG_HW + row * IMG_W + col] = (_Float16)d.e[i];
    }
  }
}

// ---------------------------------------------------------------------------
// 5) Erosion column pass via WMMA: colSum = Band(16x32) x RowSum(32x16);
//    eroded = (colSum == winW*winH).  Halo staging global->LDS uses
//    GLOBAL_LOAD_ASYNC_TO_LDS_B128 (ASYNCcnt), boundary rows zero-filled.
// ---------------------------------------------------------------------------
__global__ __launch_bounds__(256) void colpass_kernel(const _Float16* __restrict__ rowSum,
                                                      unsigned char* __restrict__ eroded) {
  __shared__ __attribute__((aligned(16))) _Float16 rsLds[64 * 256]; // rows r0-16..r0+47
  const int stripe = blockIdx.x, img = blockIdx.y;
  const int r0 = stripe * 16;

  // zero-fill out-of-range halo rows with plain LDS stores
  for (int i = threadIdx.x; i < 64 * 256; i += 256) {
    int grow = r0 - 16 + (i >> 8);
    if (grow < 0 || grow >= 256) rsLds[i] = (_Float16)0.0f;
  }
  // async-copy valid rows: 2048 16-byte units, 256 threads x 8 units.
  // LDS dest address = low 32 bits of generic pointer (AS3 offset).
  const unsigned ldsBase = (unsigned)(unsigned long long)(void*)rsLds;
  const _Float16* gbase = rowSum + (size_t)img * IMG_HW;
  {
    int u = threadIdx.x;
    for (int rep = 0; rep < 8; ++rep, u += 256) {
      int r = u >> 5;                              // 32 x 16B units per 512B row
      int grow = r0 - 16 + r;
      if (grow >= 0 && grow < 256) {
        unsigned lds_off = ldsBase + (unsigned)(u * 16);
        unsigned gv_off  = (unsigned)((grow * IMG_W + (u & 31) * 8) * 2);
        asm volatile("global_load_async_to_lds_b128 %0, %1, %2"
                     :: "v"(lds_off), "v"(gv_off), "s"(gbase)
                     : "memory");
      }
    }
  }
  asm volatile("s_wait_asynccnt 0x0" ::: "memory");
  __syncthreads();

  const int lane = threadIdx.x & 31, wid = threadIdx.x >> 5;
  const int n = lane & 15, half = lane >> 4;
  const int m = lane & 15;                        // A row = output row within tile
  F16x16 a0, a1;                                  // constant band, A layout K mapping
#pragma unroll
  for (int e = 0; e < 16; ++e) {
    int k = e + 16 * half;
    a0.e[e] = ((unsigned)(k - m - 11) <= 10u) ? (_Float16)1.0f : (_Float16)0.0f;  // rows r0-16..r0+15
    a1.e[e] = ((unsigned)(k - m + 21) <= 10u) ? (_Float16)1.0f : (_Float16)0.0f;  // rows r0+16..r0+47
  }
  for (int tile = wid * 2; tile < wid * 2 + 2; ++tile) {
    const int c0 = tile * 16;
    F16x16 b0, b1;                                // B = rowSum data, K = e + 16*half
#pragma unroll
    for (int e = 0; e < 16; ++e) {
      int K = e + 16 * half;
      b0.e[e] = rsLds[K * 256 + c0 + n];
      b1.e[e] = rsLds[(32 + K) * 256 + c0 + n];
    }
    v8f c = {};
    c = __builtin_amdgcn_wmma_f32_16x16x32_f16(false, a0.v, false, b0.v, (short)0, c, false, false);
    c = __builtin_amdgcn_wmma_f32_16x16x32_f16(false, a1.v, false, b1.v, (short)0, c, false, false);
    F32x8 d; d.v = c;
#pragma unroll
    for (int i = 0; i < 8; ++i) {
      int row = r0 + 8 * half + i;
      int col = c0 + n;
      int winW = min(col, 5) + min(255 - col, 5) + 1;
      int winH = min(row, 5) + min(255 - row, 5) + 1;
      int s = (int)(d.e[i] + 0.5f);
      eroded[(size_t)img * IMG_HW + row * IMG_W + col] = (s == winW * winH) ? 1 : 0;
    }
  }
}

// ---------------------------------------------------------------------------
// 6) FG / BG score cuts: deterministic two-level top-100 threshold
//    select iff key16 >= cut ; cut = 0 means "select all candidates"
// ---------------------------------------------------------------------------
__global__ __launch_bounds__(256) void cuts_kernel(const float* __restrict__ x,
                                                   const unsigned char* __restrict__ eroded,
                                                   const int* __restrict__ bgCutWS,
                                                   int* __restrict__ cutFWS,
                                                   int* __restrict__ cutBWS) {
  __shared__ unsigned h[256];
  __shared__ int sT, sRem, sCut;
  const int img = blockIdx.x, t = threadIdx.x;
  const float* cam = x + (size_t)img * IMG_HW;
  const unsigned char* er = eroded + (size_t)img * IMG_HW;
  const int bgCut = bgCutWS[img];

  // ---------------- FG: candidates = eroded mask ----------------
  h[t] = 0u; __syncthreads();
  for (int p = t; p < IMG_HW; p += 256)
    if (er[p]) atomicAdd(&h[keyF(img, p) >> 8], 1u);
  __syncthreads();
  if (t == 0) {
    unsigned run = 0, above = 0; int T = -1;
    for (int b = 255; b >= 0; --b) { run += h[b]; if (run >= K_FG) { T = b; above = run - h[b]; break; } }
    if (T < 0) sCut = 0; else { sT = T; sRem = (int)(K_FG - above); sCut = -1; }
  }
  __syncthreads();
  if (sCut < 0) {                                   // block-uniform branch
    h[t] = 0u; __syncthreads();
    for (int p = t; p < IMG_HW; p += 256) {
      if (!er[p]) continue;
      int k = keyF(img, p);
      if ((k >> 8) == sT) atomicAdd(&h[k & 255], 1u);
    }
    __syncthreads();
    if (t == 0) {
      unsigned run = 0; int t2 = 0;
      for (int s = 255; s >= 0; --s) { run += h[s]; if (run >= (unsigned)sRem) { t2 = s; break; } }
      sCut = (sT << 8) | t2;
    }
    __syncthreads();
  }
  if (t == 0) cutFWS[img] = sCut;
  __syncthreads();

  // ---------------- BG: candidates = camKey16 <= bgCut ----------------
  h[t] = 0u; __syncthreads();
  for (int p = t; p < IMG_HW; p += 256)
    if (camKey16(cam[p]) <= bgCut) atomicAdd(&h[keyB(img, p) >> 8], 1u);
  __syncthreads();
  if (t == 0) {
    unsigned run = 0, above = 0; int T = -1;
    for (int b = 255; b >= 0; --b) { run += h[b]; if (run >= K_FG) { T = b; above = run - h[b]; break; } }
    if (T < 0) sCut = 0; else { sT = T; sRem = (int)(K_FG - above); sCut = -1; }
  }
  __syncthreads();
  if (sCut < 0) {
    h[t] = 0u; __syncthreads();
    for (int p = t; p < IMG_HW; p += 256) {
      if (camKey16(cam[p]) > bgCut) continue;
      int k = keyB(img, p);
      if ((k >> 8) == sT) atomicAdd(&h[k & 255], 1u);
    }
    __syncthreads();
    if (t == 0) {
      unsigned run = 0; int t2 = 0;
      for (int s = 255; s >= 0; --s) { run += h[s]; if (run >= (unsigned)sRem) { t2 = s; break; } }
      sCut = (sT << 8) | t2;
    }
    __syncthreads();
  }
  if (t == 0) cutBWS[img] = sCut;
}

// ---------------------------------------------------------------------------
// 7) Final: scatter fg/bg bits into 64KB LDS mask, 3x3 clipped dilation,
//    conflict resolution, int32 labels {1, 0, -255}
// ---------------------------------------------------------------------------
__global__ __launch_bounds__(256) void final_kernel(const float* __restrict__ x,
                                                    const unsigned char* __restrict__ eroded,
                                                    const int* __restrict__ badWS,
                                                    const int* __restrict__ bgCutWS,
                                                    const int* __restrict__ cutFWS,
                                                    const int* __restrict__ cutBWS,
                                                    int* __restrict__ out) {
  extern __shared__ unsigned char mask[];          // 65536 bytes: bit0=fg seed, bit1=bg seed
  const int img = blockIdx.x, t = threadIdx.x;
  const float* cam = x + (size_t)img * IMG_HW;
  const unsigned char* er = eroded + (size_t)img * IMG_HW;
  const int bad = badWS[img];
  const int bgCut = bgCutWS[img], cutF = cutFWS[img], cutB = cutBWS[img];
  for (int p = t; p < IMG_HW; p += 256) {
    unsigned char msk = 0;
    if (!bad) {
      if (er[p] && keyF(img, p) >= cutF) msk |= 1;
      if (camKey16(cam[p]) <= bgCut && keyB(img, p) >= cutB) msk |= 2;
    }
    mask[p] = msk;
  }
  __syncthreads();
  int* ob = out + (size_t)img * IMG_HW;
  for (int p = t; p < IMG_HW; p += 256) {
    int r = p >> 8, c = p & 255;
    int acc = 0;
    for (int dr = -1; dr <= 1; ++dr) {
      int nr = r + dr; if (nr < 0 || nr > 255) continue;
      for (int dc = -1; dc <= 1; ++dc) {
        int nc = c + dc; if (nc < 0 || nc > 255) continue;
        acc |= mask[(nr << 8) | nc];
      }
    }
    int fgd = acc & 1, bgd = (acc >> 1) & 1;
    int v = -255;
    if (fgd && !bgd) v = 1;
    else if (bgd && !fgd) v = 0;
    ob[p] = v;
  }
}

// ---------------------------------------------------------------------------
extern "C" void kernel_launch(void* const* d_in, const int* in_sizes, int n_in,
                              void* d_out, int out_size, void* d_ws, size_t ws_size,
                              hipStream_t stream) {
  (void)in_sizes; (void)n_in; (void)out_size; (void)ws_size;
  const float* x = (const float*)d_in[0];
  int* out = (int*)d_out;
  char* ws = (char*)d_ws;

  // workspace layout (~50 MB)
  unsigned*      histWS   = (unsigned*)(ws + 0);            // 256*256*4 = 262144
  int*           badWS    = (int*)     (ws + 262144);       // 1024
  float*         thWS     = (float*)   (ws + 263168);       // 1024
  int*           bgCutWS  = (int*)     (ws + 264192);       // 1024
  int*           cutFWS   = (int*)     (ws + 265216);       // 1024
  int*           cutBWS   = (int*)     (ws + 266240);       // 1024
  _Float16*      rowSumWS = (_Float16*)(ws + 267264);                   // 32 MB (16B aligned)
  unsigned char* erodedWS = (unsigned char*)(ws + 267264 + 33554432);   // 16 MB

  hist_kernel  <<<256, 256, 0, stream>>>(x, histWS, badWS);
  otsu_kernel  <<<1,   256, 0, stream>>>(histWS, thWS);
  bgcut_kernel <<<256, 256, 0, stream>>>(x, histWS, bgCutWS);
  rowpass_kernel<<<dim3(16, 256), 256, 16 * 304 * 2, stream>>>(x, thWS, rowSumWS);
  colpass_kernel<<<dim3(16, 256), 256, 0, stream>>>(rowSumWS, erodedWS);
  cuts_kernel  <<<256, 256, 0, stream>>>(x, erodedWS, bgCutWS, cutFWS, cutBWS);
  final_kernel <<<256, 256, 65536, stream>>>(x, erodedWS, badWS, bgCutWS, cutFWS, cutBWS, out);
}
